// BatchConv1d_51204600103652
// MI455X (gfx1250) — matrix operand order repaired and checked by reading, not compiled, not run on hardware
//
#include <hip/hip_runtime.h>
#include <hip/hip_bf16.h>
#include <stdint.h>

// ---- problem constants (fixed by the reference) ----
#define KW    3
#define BQ    8
#define LQ    1024
#define LK    1024
#define DD    512          // q feature dim (reduction dim of stage 1)
#define CC    512          // k feature dim (reduction dim of stage 2)
#define NKC   (CC * KW)    // 1536
#define MROWS (BQ * LQ)    // 8192

typedef __attribute__((ext_vector_type(16))) __bf16 v16bf;
typedef __attribute__((ext_vector_type(8)))  float  v8f;
typedef int v4i __attribute__((vector_size(16)));   // matches async builtin's pointee type

union Frag16 { uint4 u[2]; v16bf v; };

// ---- CDNA5 async global->LDS path (guarded; falls back to ds_store staging) ----
#if defined(__gfx1250__) && \
    __has_builtin(__builtin_amdgcn_global_load_async_to_lds_b128) && \
    __has_builtin(__builtin_amdgcn_s_wait_asynccnt)
#define HAVE_ASYNC_LDS 1
#else
#define HAVE_ASYNC_LDS 0
#endif

static __device__ __forceinline__ void cp_b128(uint16_t* dst_lds, const uint16_t* src_glb) {
#if HAVE_ASYNC_LDS
  __builtin_amdgcn_global_load_async_to_lds_b128((v4i*)src_glb, (v4i*)dst_lds, 0, 0);
#else
  *(uint4*)dst_lds = *(const uint4*)src_glb;
#endif
}

static __device__ __forceinline__ void async_stage_wait() {
#if HAVE_ASYNC_LDS
  __builtin_amdgcn_s_wait_asynccnt(0);
#endif
}

static __device__ __forceinline__ uint16_t f2bf(float f) {
  uint32_t u = __float_as_uint(f);
  uint32_t r = u + 0x7FFFu + ((u >> 16) & 1u);   // round-to-nearest-even
  return (uint16_t)(r >> 16);
}

// ---------------- elementwise fp32 -> bf16 ----------------
__global__ void cvt_f32_bf16(const float* __restrict__ src,
                             uint16_t* __restrict__ dst, int n) {
  int i = blockIdx.x * blockDim.x + threadIdx.x;
  if (i < n) dst[i] = f2bf(src[i]);
}

// ---- transpose W_kernel (512 x 1536) -> WkT (1536 x 512) with the
// ---- (C,KW) de-interleave folded in: row n' = w*512 + c. Also permute b_kernel.
__global__ void prep_w(const float* __restrict__ Wk, const float* __restrict__ bk,
                       uint16_t* __restrict__ WkT, float* __restrict__ bkp) {
  int idx = blockIdx.x * blockDim.x + threadIdx.x;  // 1536*512 threads
  int kk = idx & (DD - 1);
  int np = idx >> 9;           // 0..1535 : np = w*512 + c
  int c  = np & (CC - 1);
  int w  = np >> 9;
  WkT[idx] = f2bf(Wk[(size_t)kk * NKC + c * KW + w]);
  if (kk == 0) bkp[np] = bk[c * KW + w];
}

// ---- bias rows: bias[i] = q[i,:]·W_bias + b_bias + bias_b (fp32, wave per row)
__global__ void bias_rows(const float* __restrict__ q, const float* __restrict__ Wb,
                          const float* __restrict__ bb, const float* __restrict__ bbg,
                          float* __restrict__ bias) {
  int row  = blockIdx.x * 8 + (threadIdx.x >> 5);
  int lane = threadIdx.x & 31;
  const float* qr = q + (size_t)row * DD;
  float s = 0.f;
  for (int c = lane; c < DD; c += 32) s += qr[c] * Wb[c];
  for (int off = 16; off > 0; off >>= 1) s += __shfl_xor(s, off, 32);
  if (lane == 0) bias[row] = s + bb[0] + bbg[0];
}

// LDS row padding: 20-dword stride -> the 16 lanes of a half-wave start on 16
// distinct bank groups (gcd(20,64)=4, period 16) => conflict-free ds_load_b128.
#define PAD32 40     // halves per 32-half row
#define AS_STRIDE 104 // halves per 96-half A-slab row (52 dwords, gcd=4, period 16)

// ---------------- stage 1: kermat(8192 x 1536, bf16, cols ordered (w,c)) ----------------
// M=8192 K=512 N=1536. Workgroup 64M x 128N, 8 waves of 16x64; double-buffered LDS.
__global__ void __launch_bounds__(256) gemm_proj(
    const uint16_t* __restrict__ qbf, const uint16_t* __restrict__ WkT,
    const float* __restrict__ bkp, uint16_t* __restrict__ kermat) {
  __shared__ uint16_t sA[2][64 * PAD32];    // 2 x  5,120 B
  __shared__ uint16_t sB[2][128 * PAD32];   // 2 x 10,240 B

  const int t      = threadIdx.x;
  const int lane   = t & 31;
  const int wv     = t >> 5;
  const int lane16 = lane & 15;
  const int half   = lane >> 4;
  const int waveM  = wv & 3;
  const int waveN  = wv >> 2;
  const int tileM  = blockIdx.x * 64;
  const int tileN  = blockIdx.y * 128;

  const uint16_t* Ag = qbf + (size_t)tileM * DD;
  const uint16_t* Bg = WkT + (size_t)tileN * DD;

  // stage(buf, kb): A 64x32 (256 chunks), B 128x32 (512 chunks)
  auto stage = [&](int buf, int kb) {
    {
      int row = t >> 2, ch = t & 3;   // exactly 256 chunks
      cp_b128(&sA[buf][row * PAD32 + ch * 8], Ag + (size_t)row * DD + kb + ch * 8);
    }
#pragma unroll
    for (int rep = 0; rep < 2; ++rep) {
      int idx = t + rep * 256;
      int row = idx >> 2, ch = idx & 3;
      cp_b128(&sB[buf][row * PAD32 + ch * 8], Bg + (size_t)row * DD + kb + ch * 8);
    }
  };

  v8f acc[4] = {};
  stage(0, 0);
  for (int it = 0; it < DD / 32; ++it) {
    const int buf = it & 1;
    async_stage_wait();
    __syncthreads();
    if (it + 1 < DD / 32) stage(1 - buf, (it + 1) * 32);

    Frag16 a;
    const uint16_t* ap = &sA[buf][(waveM * 16 + lane16) * PAD32 + half * 8];
    a.u[0] = *(const uint4*)ap;
    a.u[1] = *(const uint4*)(ap + 16);
#pragma unroll
    for (int nf = 0; nf < 4; ++nf) {
      const uint16_t* bp = &sB[buf][(waveN * 64 + nf * 16 + lane16) * PAD32 + half * 8];
      Frag16 bf;
      bf.u[0] = *(const uint4*)bp;
      bf.u[1] = *(const uint4*)(bp + 16);
      acc[nf] = __builtin_amdgcn_wmma_f32_16x16x32_bf16(
          false, a.v, false, bf.v, (short)0, acc[nf], false, false);
    }
    __syncthreads();
  }
#pragma unroll
  for (int nf = 0; nf < 4; ++nf) {
    int   col = tileN + waveN * 64 + nf * 16 + lane16;
    float bia = bkp[col];
#pragma unroll
    for (int v = 0; v < 8; ++v) {
      int row = tileM + waveM * 16 + half * 8 + v;
      kermat[(size_t)row * NKC + col] = f2bf(acc[nf][v] + bia);
    }
  }
}

// ---------------- stage 2: out[b,i,j] = Σ_w Σ_c ker[b,i,w,c]·k[b,j+w-1,c] + bias[b,i] ----
// Per batch: M=1024 N=1024 K=3*512. Workgroup 64M x 128N, 8 waves of 16x64.
// Double-buffered LDS: 130x32 k-slab covers all 3 window shifts; A-slab 64x96 covers all 3 w.
__global__ void __launch_bounds__(256) conv_attn(
    const uint16_t* __restrict__ kermat, const uint16_t* __restrict__ kbf,
    const float* __restrict__ bias, float* __restrict__ out) {
  __shared__ uint16_t sB[2][132 * PAD32];     // 2 x 10,560 B
  __shared__ uint16_t sA[2][64 * AS_STRIDE];  // 2 x 13,312 B

  const int t      = threadIdx.x;
  const int lane   = t & 31;
  const int wv     = t >> 5;
  const int lane16 = lane & 15;
  const int half   = lane >> 4;
  const int waveM  = wv & 3;        // 0..3  (16-row strip)
  const int waveN  = wv >> 2;       // 0..1  (64-col strip)
  const int b      = blockIdx.z;
  const int tileM  = blockIdx.y * 64;
  const int tileN  = blockIdx.x * 128;

  const uint16_t* Kb = kbf + (size_t)b * LK * CC;
  const uint16_t* Ab = kermat + (size_t)(b * LQ + tileM) * NKC;

  // stage(buf, cb): k-slab rows tileN-1..tileN+128 (130x32, zero-pad OOB) + A-slab 64x96
  auto stage = [&](int buf, int cb) {
    for (int idx = t; idx < 130 * 4; idx += 256) {
      int row = idx >> 2, ch = idx & 3;
      uint16_t* dst = &sB[buf][row * PAD32 + ch * 8];
      int gr = tileN - 1 + row;
      if ((unsigned)gr < (unsigned)LK)
        cp_b128(dst, Kb + (size_t)gr * CC + cb + ch * 8);
      else
        *(uint4*)dst = make_uint4(0u, 0u, 0u, 0u);
    }
#pragma unroll
    for (int rep = 0; rep < 3; ++rep) {       // 64*12 = 768 chunks
      int idx = t + rep * 256;
      int row = idx / 12, cc = idx - row * 12;
      cp_b128(&sA[buf][row * AS_STRIDE + cc * 8],
              Ab + (size_t)row * NKC + (cc >> 2) * CC + cb + (cc & 3) * 8);
    }
  };

  v8f acc[4] = {};
  stage(0, 0);
  for (int it = 0; it < CC / 32; ++it) {
    const int buf = it & 1;
    async_stage_wait();
    __syncthreads();
    if (it + 1 < CC / 32) stage(1 - buf, (it + 1) * 32);

#pragma unroll
    for (int w = 0; w < KW; ++w) {
      Frag16 a;
      const uint16_t* ap = &sA[buf][(waveM * 16 + lane16) * AS_STRIDE + w * 32 + half * 8];
      a.u[0] = *(const uint4*)ap;
      a.u[1] = *(const uint4*)(ap + 16);
#pragma unroll
      for (int nf = 0; nf < 4; ++nf) {
        int jl = waveN * 64 + nf * 16 + lane16;       // local output column
        const uint16_t* bp = &sB[buf][(jl + w) * PAD32 + half * 8];  // row 0 == tileN-1
        Frag16 bf;
        bf.u[0] = *(const uint4*)bp;
        bf.u[1] = *(const uint4*)(bp + 16);
        acc[nf] = __builtin_amdgcn_wmma_f32_16x16x32_bf16(
            false, a.v, false, bf.v, (short)0, acc[nf], false, false);
      }
    }
    __syncthreads();
  }

  float* obase = out + (size_t)b * LQ * LK;
#pragma unroll
  for (int v = 0; v < 8; ++v) {
    int   row = tileM + waveM * 16 + half * 8 + v;
    float bi  = bias[b * LQ + row];
#pragma unroll
    for (int nf = 0; nf < 4; ++nf)
      obase[(size_t)row * LK + tileN + waveN * 64 + nf * 16 + lane16] = acc[nf][v] + bi;
  }
}

extern "C" void kernel_launch(void* const* d_in, const int* in_sizes, int n_in,
                              void* d_out, int out_size, void* d_ws, size_t ws_size,
                              hipStream_t stream) {
  const float* q   = (const float*)d_in[0];   // (8,1024,512)
  const float* k   = (const float*)d_in[1];   // (8,1024,512)
  const float* Wk  = (const float*)d_in[2];   // (512,1536)
  const float* bk  = (const float*)d_in[3];   // (1536,)
  const float* Wb  = (const float*)d_in[4];   // (512,1)
  const float* bb  = (const float*)d_in[5];   // (1,)
  const float* bbg = (const float*)d_in[6];   // (1,)
  float* out = (float*)d_out;                 // (8,1024,1024) fp32

  // workspace layout (all 256B aligned): total ~43.6 MB
  char* ws = (char*)d_ws;
  uint16_t* qbf    = (uint16_t*)(ws);                       //  8,388,608 B
  uint16_t* kbf    = (uint16_t*)(ws + 8388608);             //  8,388,608 B
  uint16_t* WkT    = (uint16_t*)(ws + 16777216);            //  1,572,864 B
  uint16_t* kermat = (uint16_t*)(ws + 18350080);            // 25,165,824 B
  float*    bkp    = (float*)   (ws + 43515904);            //      6,144 B
  float*    bias   = (float*)   (ws + 43522048);            //     32,768 B

  const int nq = MROWS * DD;   // 4,194,304
  cvt_f32_bf16<<<nq / 256, 256, 0, stream>>>(q, qbf, nq);
  cvt_f32_bf16<<<nq / 256, 256, 0, stream>>>(k, kbf, nq);
  prep_w<<<(NKC * DD) / 256, 256, 0, stream>>>(Wk, bk, WkT, bkp);
  bias_rows<<<MROWS / 8, 256, 0, stream>>>(q, Wb, bb, bbg, bias);

  gemm_proj<<<dim3(MROWS / 64, NKC / 128), 256, 0, stream>>>(qbf, WkT, bkp, kermat);
  conv_attn<<<dim3(LK / 128, LQ / 64, BQ), 256, 0, stream>>>(kermat, kbf, bias, out);
}
